// DenseDilatedKnnGraph_30124900614634
// MI455X (gfx1250) — matrix operand order repaired and checked by reading, not compiled, not run on hardware
//
#include <hip/hip_runtime.h>

typedef _Float16 v16h __attribute__((ext_vector_type(16)));
typedef _Float16 v8h  __attribute__((ext_vector_type(8)));
typedef float    v8f  __attribute__((ext_vector_type(8)));

#define BATCH 4
#define CH    64
#define NPTS  8192
#define KTOT  32          // top-k before dilation
#define KOUT  16          // after dilation 2

// ---------------------------------------------------------------------------
// Kernel 1: L2-normalize along channel dim and transpose (B,C,N) -> f16 (B,N,C)
// ---------------------------------------------------------------------------
__global__ void __launch_bounds__(256)
norm_transpose_kernel(const float* __restrict__ in, _Float16* __restrict__ out) {
    int t = blockIdx.x * blockDim.x + threadIdx.x;   // t = b*NPTS + n
    if (t >= BATCH * NPTS) return;
    int b = t / NPTS;
    int n = t % NPTS;
    const float* p = in + (size_t)b * CH * NPTS + n;

    float vals[CH];
    float s = 0.f;
#pragma unroll
    for (int c = 0; c < CH; ++c) {
        float v = p[(size_t)c * NPTS];   // coalesced across consecutive n
        vals[c] = v;
        s += v * v;
    }
    float scale = 1.f / fmaxf(sqrtf(s), 1e-12f);

    _Float16* q = out + (size_t)t * CH;
#pragma unroll
    for (int c = 0; c < CH; ++c) q[c] = (_Float16)(vals[c] * scale);
}

// ---------------------------------------------------------------------------
// Kernel 2: fused WMMA gram-tile + per-row top-32 (largest dot == smallest
// distance for unit vectors) + dilated edge_index write.
// One wave per (batch, 16-row tile): 4*512 = 2048 blocks.
// Fast path: compare WMMA accumulators against register-cached per-row
// thresholds; a wave-uniform ballot skips the LDS dump + scan on reject tiles.
// ---------------------------------------------------------------------------
__global__ void __launch_bounds__(32)
knn_wmma_kernel(const _Float16* __restrict__ xh,
                const _Float16* __restrict__ yh,
                int* __restrict__ out) {
    __shared__ float tile[16][17];        // 16x16 dot tile (+pad)
    __shared__ float best_d[16][KTOT];    // descending dot values per row
    __shared__ int   best_i[16][KTOT];
    __shared__ float worst_s[16];         // best_d[row][KTOT-1] mirror

    const int lane = threadIdx.x;         // 0..31
    const int b    = blockIdx.y;
    const int m0   = blockIdx.x * 16;

    const int m = lane & 15;              // row / column within tile
    const int g = lane >> 4;              // lane half-group

    if (lane < 16) {
#pragma unroll 1
        for (int k = 0; k < KTOT; ++k) { best_d[lane][k] = -3.0e38f; best_i[lane][k] = 0; }
        worst_s[lane] = -3.0e38f;
    }

    // ---- A matrix: rows m0..m0+15, all 64 K, ISA 16-bit A layout ----
    // group g holds K chunks [g*8, g*8+8) and [16+g*8, 16+g*8+8) per 32-K slice
    const _Float16* xrow = xh + ((size_t)b * NPTS + m0 + m) * CH;
    v16h a0, a1;
    *((v8h*)&a0)       = *(const v8h*)(xrow +  0 + g * 8);
    *(((v8h*)&a0) + 1) = *(const v8h*)(xrow + 16 + g * 8);
    *((v8h*)&a1)       = *(const v8h*)(xrow + 32 + g * 8);
    *(((v8h*)&a1) + 1) = *(const v8h*)(xrow + 48 + g * 8);

    const _Float16* ybase = yh + (size_t)b * NPTS * CH;

    __syncthreads();

    // register cache of the 32nd-best dot for this lane's 8 accumulator rows
    float worstReg[8];
#pragma unroll
    for (int r = 0; r < 8; ++r) worstReg[r] = worst_s[r + 8 * g];

    for (int nt = 0; nt < NPTS / 16; ++nt) {
        const int n0 = nt * 16;
        // ---- B matrix: column n = lane%16 is row (n0+n) of yf; group g holds
        //      K chunks [g*8, g*8+8) and [16+g*8, 16+g*8+8) per 32-K slice ----
        const _Float16* yrow = ybase + (size_t)(n0 + m) * CH;
        v16h b0, b1;
        *((v8h*)&b0)       = *(const v8h*)(yrow +  0 + g * 8);
        *(((v8h*)&b0) + 1) = *(const v8h*)(yrow + 16 + g * 8);
        *((v8h*)&b1)       = *(const v8h*)(yrow + 32 + g * 8);
        *(((v8h*)&b1) + 1) = *(const v8h*)(yrow + 48 + g * 8);

        // prefetch next tile's y rows toward the WGP (global_prefetch_b8)
        if (nt + 1 < NPTS / 16)
            __builtin_prefetch(yrow + 16 * CH, 0, 3);

        v8f c = {};
        c = __builtin_amdgcn_wmma_f32_16x16x32_f16(false, a0, false, b0,
                                                   (short)0, c, false, false);
        c = __builtin_amdgcn_wmma_f32_16x16x32_f16(false, a1, false, b1,
                                                   (short)0, c, false, false);

        // ---- fast reject: does any accumulator beat its row's threshold? ----
        bool cand = false;
#pragma unroll
        for (int r = 0; r < 8; ++r) cand |= (c[r] > worstReg[r]);

        if (__builtin_amdgcn_ballot_w32(cand) != 0u) {   // wave-uniform branch
            // C layout: element r of v8f is (M = r + 8*g, N = lane%16)
#pragma unroll
            for (int r = 0; r < 8; ++r) tile[r + 8 * g][m] = c[r];
            __syncthreads();

            // lanes 0..15: serial scan of 16 candidates for own row; strict
            // comparisons keep lower-index-first tie behavior like jax top_k
            if (lane < 16) {
                float worst = worst_s[lane];
#pragma unroll 1
                for (int j = 0; j < 16; ++j) {
                    float v = tile[lane][j];
                    if (v > worst) {
                        int pos = KTOT - 1;
                        while (pos > 0 && best_d[lane][pos - 1] < v) {
                            best_d[lane][pos] = best_d[lane][pos - 1];
                            best_i[lane][pos] = best_i[lane][pos - 1];
                            --pos;
                        }
                        best_d[lane][pos] = v;
                        best_i[lane][pos] = n0 + j;
                        worst = best_d[lane][KTOT - 1];
                    }
                }
                worst_s[lane] = worst;
            }
            __syncthreads();

            // thresholds may have moved: refresh register cache
#pragma unroll
            for (int r = 0; r < 8; ++r) worstReg[r] = worst_s[r + 8 * g];
        }
    }

    // ---- write dilated edge_index: out layout (2, B, N, KOUT) + leading 0 ----
    if (lane < 16) {
        const int row   = m0 + lane;
        size_t base_nn  = 1 + (((size_t)0 * BATCH + b) * NPTS + row) * KOUT;
        size_t base_ctr = 1 + (((size_t)1 * BATCH + b) * NPTS + row) * KOUT;
#pragma unroll 1
        for (int k = 0; k < KOUT; ++k) {
            out[base_nn  + k] = best_i[lane][2 * k];  // every 2nd neighbor
            out[base_ctr + k] = row;
        }
    }
    if (b == 0 && blockIdx.x == 0 && lane == 0) out[0] = 0;  // scalar output
}

// ---------------------------------------------------------------------------
extern "C" void kernel_launch(void* const* d_in, const int* in_sizes, int n_in,
                              void* d_out, int out_size, void* d_ws, size_t ws_size,
                              hipStream_t stream) {
    const float* x = (const float*)d_in[0];
    const float* y = (const float*)d_in[1];
    int* out = (int*)d_out;

    _Float16* xh = (_Float16*)d_ws;                               // 4 MB
    _Float16* yh = xh + (size_t)BATCH * NPTS * CH;                // +4 MB

    const int total = BATCH * NPTS;
    dim3 nblk((total + 255) / 256), nthr(256);
    norm_transpose_kernel<<<nblk, nthr, 0, stream>>>(x, xh);
    norm_transpose_kernel<<<nblk, nthr, 0, stream>>>(y, yh);

    dim3 grid(NPTS / 16, BATCH), block(32);
    knn_wmma_kernel<<<grid, block, 0, stream>>>(xh, yh, out);
}
